// SCAM_6210522710373
// MI455X (gfx1250) — matrix operand order repaired and checked by reading, not compiled, not run on hardware
//
#include <hip/hip_runtime.h>

typedef __attribute__((ext_vector_type(16))) __bf16 v16bf;
typedef __attribute__((ext_vector_type(8)))  __bf16 v8bf;
typedef __attribute__((ext_vector_type(8)))  float  v8f;

#define B_ 8
#define C_ 128
#define H_ 180
#define W_ 320
#define HW_ (H_*W_)
#define NTILES (W_/16)          /* 20 key/query tiles per row */
#define NWAVES 4
#define SCALE_QK 0.08838834764831845f  /* 128^-0.5 */

/* ---- LDS layout (bytes) ---- */
#define KT_OFF     0
#define KT_BYTES   (C_*W_*2)                   /* K^T: 128 x 320 bf16 = 80KB */
#define V_OFF      (KT_OFF + KT_BYTES)
#define V_BYTES    (W_*C_*2)                   /* V: 320 x 128 bf16 = 80KB  */
#define STG_OFF    (V_OFF + V_BYTES)
#define STG_WB     (16*C_*2)                   /* per-wave 16x128 bf16 stage */
#define SBUF_OFF   (STG_OFF + NWAVES*STG_WB)
#define SBUF_WB    (16*W_*4)                   /* per-wave 16x320 f32 scores */
#define PBUF_OFF   (SBUF_OFF + NWAVES*SBUF_WB)
#define PBUF_WB    (16*W_*2)                   /* per-wave 16x320 bf16 probs */
#define SMEM_BYTES (PBUF_OFF + NWAVES*PBUF_WB) /* 303104 B <= 320KB WGP LDS */

/* A-fragment (16xK tile, 16-bit): lane m = L%16, half = L/16.
   elems 0..7  -> K = kbase + half*8 + i
   elems 8..15 -> K = kbase + 16 + half*8 + i      (ISA 7.12.2) */
__device__ __forceinline__ v16bf frag_a(const __bf16* p, int stride, int lane) {
    int m = lane & 15, hf = lane >> 4;
    const __bf16* r = p + m * stride + hf * 8;
    v8bf lo = *(const v8bf*)(r);
    v8bf hi = *(const v8bf*)(r + 16);
    return __builtin_shufflevector(lo, hi, 0,1,2,3,4,5,6,7,8,9,10,11,12,13,14,15);
}

/* B-fragment (32x16 tile, 16-bit): lane n = L%16, half = L/16;
   elems 0..7 -> K = half*8 + i, elems 8..15 -> K = 16 + half*8 + i */
__device__ __forceinline__ v16bf frag_b(const __bf16* p, int stride, int lane) {
    int nn = lane & 15, hf = lane >> 4;
    const __bf16* r = p + nn + hf * 8 * stride;
    v16bf v;
    #pragma unroll
    for (int i = 0; i < 8; ++i) {
        v[i] = r[i * stride];
        v[i + 8] = r[(16 + i) * stride];
    }
    return v;
}

__device__ __forceinline__ v8f wmma_bf16(v16bf a, v16bf b, v8f c) {
    return __builtin_amdgcn_wmma_f32_16x16x32_bf16(false, a, false, b,
                                                   (short)0, c, false, false);
}

/* Transpose projection weights (o,i)->(i,o) and narrow to bf16 in workspace */
__global__ void scam_wt_prep(const float* __restrict__ wq,
                             const float* __restrict__ wk,
                             const float* __restrict__ wv,
                             __bf16* __restrict__ wt) {
    int i = blockIdx.x * blockDim.x + threadIdx.x;
    if (i >= 3 * C_ * C_) return;
    int m = i / (C_ * C_), r = i % (C_ * C_);
    int o = r / C_, c = r % C_;
    const float* w = (m == 0) ? wq : ((m == 1) ? wk : wv);
    wt[m * C_ * C_ + c * C_ + o] = (__bf16)w[o * C_ + c];
}

__global__ __launch_bounds__(128)
void scam_main(const float* __restrict__ x_l, const float* __restrict__ x_r,
               const float* __restrict__ g_l, const float* __restrict__ b_l,
               const float* __restrict__ g_r, const float* __restrict__ b_r,
               const __bf16* __restrict__ wt,
               const float* __restrict__ b_q, const float* __restrict__ b_k,
               const float* __restrict__ b_v,
               const float* __restrict__ beta, const float* __restrict__ gamma,
               float* __restrict__ out)
{
    extern __shared__ char smem[];
    __bf16* KT = (__bf16*)(smem + KT_OFF);   /* KT[c][p] */
    __bf16* Vb = (__bf16*)(smem + V_OFF);    /* V[p][c]  */
    const int tid = threadIdx.x;
    const int lane = tid & 31, wv = tid >> 5;
    __bf16* stg  = (__bf16*)(smem + STG_OFF + wv * STG_WB);
    float*  Sbuf = (float*) (smem + SBUF_OFF + wv * SBUF_WB);
    __bf16* Pbuf = (__bf16*)(smem + PBUF_OFF + wv * PBUF_WB);

    const int rb = blockIdx.x;
    const int d  = rb & 1;          /* 0: r2l (delta_l), 1: l2r (delta_r) */
    const int rh = rb >> 1;
    const int b  = rh / H_, h = rh % H_;

    const float* xq = d ? x_r : x_l;   const float* xk = d ? x_l : x_r;
    const float* gq = d ? g_r : g_l;   const float* bqln = d ? b_r : b_l;
    const float* gk = d ? g_l : g_r;   const float* bkln = d ? b_l : b_r;
    const float* bg = d ? gamma : beta;
    float* outp = out + (size_t)d * ((size_t)B_ * C_ * HW_)
                      + (size_t)b * C_ * HW_ + (size_t)h * W_;

    const __bf16* WqT = wt;
    const __bf16* WkT = wt + C_ * C_;
    const __bf16* WvT = wt + 2 * C_ * C_;

    const size_t xbase = (size_t)b * C_ * HW_ + (size_t)h * W_;
    const int pl = lane & 15, seg = lane >> 4;
    const int n = pl, m0 = seg << 3;   /* D-matrix: n = L%16, m = j + 8*(L/16) */

    /* ================= Phase A: build K^T and V for the key side ========= */
    for (int t = wv; t < NTILES; t += NWAVES) {
        const int pbase = t * 16;
        /* stage raw pixels (bf16) + LN stats, coalesced along w */
        float s1 = 0.f, s2 = 0.f;
        for (int ci = 0; ci < 64; ++ci) {
            int c = seg * 64 + ci;
            float v = xk[xbase + (size_t)c * HW_ + pbase + pl];
            s1 += v; s2 += v * v;
            stg[pl * C_ + c] = (__bf16)v;
        }
        s1 += __shfl_xor(s1, 16, 32);
        s2 += __shfl_xor(s2, 16, 32);
        float mean = s1 * (1.f / C_);
        float var  = s2 * (1.f / C_) - mean * mean;
        float rstd = rsqrtf(var + 1e-6f);

        /* V projection from raw x */
        {
            v8f acc[8] = {};
            for (int ks = 0; ks < 4; ++ks) {
                v16bf a = frag_a(stg + ks * 32, C_, lane);
                #pragma unroll
                for (int ot = 0; ot < 8; ++ot) {
                    v16bf bb = frag_b(WvT + ks * 32 * C_ + ot * 16, C_, lane);
                    acc[ot] = wmma_bf16(a, bb, acc[ot]);
                }
            }
            #pragma unroll
            for (int ot = 0; ot < 8; ++ot) {
                float bias = b_v[ot * 16 + n];
                #pragma unroll
                for (int j = 0; j < 8; ++j)
                    Vb[(pbase + m0 + j) * C_ + ot * 16 + n] =
                        (__bf16)(acc[ot][j] + bias);
            }
        }
        /* normalize stage in place */
        for (int ci = 0; ci < 64; ++ci) {
            int c = seg * 64 + ci;
            float v = (float)stg[pl * C_ + c];
            stg[pl * C_ + c] = (__bf16)((v - mean) * rstd * gk[c] + bkln[c]);
        }
        /* K projection -> K^T (transposed store for score B-fragments) */
        {
            v8f acc[8] = {};
            for (int ks = 0; ks < 4; ++ks) {
                v16bf a = frag_a(stg + ks * 32, C_, lane);
                #pragma unroll
                for (int ot = 0; ot < 8; ++ot) {
                    v16bf bb = frag_b(WkT + ks * 32 * C_ + ot * 16, C_, lane);
                    acc[ot] = wmma_bf16(a, bb, acc[ot]);
                }
            }
            #pragma unroll
            for (int ot = 0; ot < 8; ++ot) {
                float bias = b_k[ot * 16 + n];
                #pragma unroll
                for (int j = 0; j < 8; ++j)
                    KT[(ot * 16 + n) * W_ + pbase + m0 + j] =
                        (__bf16)(acc[ot][j] + bias);
            }
        }
    }
    __syncthreads();

    /* ================= Phase B: Q-proj + attention per query tile ======== */
    for (int t = wv; t < NTILES; t += NWAVES) {
        const int pbase = t * 16;
        float s1 = 0.f, s2 = 0.f;
        for (int ci = 0; ci < 64; ++ci) {
            int c = seg * 64 + ci;
            float v = xq[xbase + (size_t)c * HW_ + pbase + pl];
            s1 += v; s2 += v * v;
            stg[pl * C_ + c] = (__bf16)v;
        }
        s1 += __shfl_xor(s1, 16, 32);
        s2 += __shfl_xor(s2, 16, 32);
        float mean = s1 * (1.f / C_);
        float var  = s2 * (1.f / C_) - mean * mean;
        float rstd = rsqrtf(var + 1e-6f);
        for (int ci = 0; ci < 64; ++ci) {
            int c = seg * 64 + ci;
            float v = (float)stg[pl * C_ + c];
            stg[pl * C_ + c] = (__bf16)((v - mean) * rstd * gq[c] + bqln[c]);
        }
        /* Q projection (accumulate fully in regs, then overwrite stage) */
        {
            v8f acc[8] = {};
            for (int ks = 0; ks < 4; ++ks) {
                v16bf a = frag_a(stg + ks * 32, C_, lane);
                #pragma unroll
                for (int ot = 0; ot < 8; ++ot) {
                    v16bf bb = frag_b(WqT + ks * 32 * C_ + ot * 16, C_, lane);
                    acc[ot] = wmma_bf16(a, bb, acc[ot]);
                }
            }
            #pragma unroll
            for (int ot = 0; ot < 8; ++ot) {
                float bias = b_q[ot * 16 + n];
                #pragma unroll
                for (int j = 0; j < 8; ++j)
                    stg[(m0 + j) * C_ + ot * 16 + n] = (__bf16)(acc[ot][j] + bias);
            }
        }
        /* scores S = (Q K^T) * C^-1/2, fp32 in LDS */
        for (int kt = 0; kt < NTILES; ++kt) {
            v8f acc = {};
            for (int ks = 0; ks < 4; ++ks) {
                v16bf a = frag_a(stg + ks * 32, C_, lane);
                v16bf bb = frag_b(KT + ks * 32 * W_ + kt * 16, W_, lane);
                acc = wmma_bf16(a, bb, acc);
            }
            #pragma unroll
            for (int j = 0; j < 8; ++j)
                Sbuf[(m0 + j) * W_ + kt * 16 + n] = acc[j] * SCALE_QK;
        }
        __syncthreads();  /* uniform: every wave runs 5 iterations */

        /* softmax along width: lane = (row, half-row) */
        {
            int row = pl, u0 = seg * 160;
            float mx = -3.0e38f;
            for (int u = u0; u < u0 + 160; ++u)
                mx = fmaxf(mx, Sbuf[row * W_ + u]);
            mx = fmaxf(mx, __shfl_xor(mx, 16, 32));
            float sum = 0.f;
            for (int u = u0; u < u0 + 160; ++u)
                sum += __expf(Sbuf[row * W_ + u] - mx);
            sum += __shfl_xor(sum, 16, 32);
            float inv = 1.f / sum;
            for (int u = u0; u < u0 + 160; ++u)
                Pbuf[row * W_ + u] = (__bf16)(__expf(Sbuf[row * W_ + u] - mx) * inv);
        }
        __syncthreads();

        /* F = P @ V  (K-dim = 320 keys, 10 steps of 32) */
        v8f facc[8] = {};
        for (int us = 0; us < 10; ++us) {
            v16bf a = frag_a(Pbuf + us * 32, W_, lane);
            #pragma unroll
            for (int ot = 0; ot < 8; ++ot) {
                v16bf bb = frag_b(Vb + us * 32 * C_ + ot * 16, C_, lane);
                facc[ot] = wmma_bf16(a, bb, facc[ot]);
            }
        }
        /* scale by beta/gamma, transpose through LDS (reuse Pbuf as 128x16 f32) */
        float* FT = (float*)Pbuf;
        #pragma unroll
        for (int ot = 0; ot < 8; ++ot) {
            float sc = bg[ot * 16 + n];
            #pragma unroll
            for (int j = 0; j < 8; ++j)
                FT[(ot * 16 + n) * 16 + m0 + j] = facc[ot][j] * sc;
        }
        __syncthreads();

        /* coalesced (B,C,H,W) store: each lane writes 64B rows of 4 channels */
        for (int cc = 0; cc < 4; ++cc) {
            int c = cc * 32 + lane;
            const float4* src = (const float4*)(FT + c * 16);
            float4 a0 = src[0], a1 = src[1], a2 = src[2], a3 = src[3];
            float4* dst = (float4*)(outp + (size_t)c * HW_ + pbase);
            dst[0] = a0; dst[1] = a1; dst[2] = a2; dst[3] = a3;
        }
        __syncthreads();
    }
}

extern "C" void kernel_launch(void* const* d_in, const int* in_sizes, int n_in,
                              void* d_out, int out_size, void* d_ws, size_t ws_size,
                              hipStream_t stream) {
    const float* x_l  = (const float*)d_in[0];
    const float* x_r  = (const float*)d_in[1];
    const float* g_l  = (const float*)d_in[2];
    const float* b_l  = (const float*)d_in[3];
    const float* g_r  = (const float*)d_in[4];
    const float* b_r  = (const float*)d_in[5];
    const float* w_q  = (const float*)d_in[6];
    const float* b_q  = (const float*)d_in[7];
    const float* w_k  = (const float*)d_in[8];
    const float* b_k  = (const float*)d_in[9];
    const float* w_v  = (const float*)d_in[10];
    const float* b_v  = (const float*)d_in[11];
    const float* beta = (const float*)d_in[12];
    const float* gam  = (const float*)d_in[13];
    __bf16* wt = (__bf16*)d_ws;   /* 3*128*128 bf16 = 96KB */

    (void)in_sizes; (void)n_in; (void)out_size; (void)ws_size;
    hipFuncSetAttribute(reinterpret_cast<const void*>(scam_main),
                        hipFuncAttributeMaxDynamicSharedMemorySize, SMEM_BYTES);

    scam_wt_prep<<<(3 * C_ * C_ + 255) / 256, 256, 0, stream>>>(w_q, w_k, w_v, wt);
    scam_main<<<2 * B_ * H_, 128, SMEM_BYTES, stream>>>(
        x_l, x_r, g_l, b_l, g_r, b_r, wt, b_q, b_k, b_v, beta, gam, (float*)d_out);
}
